// HistoryEncoder_22720376996579
// MI455X (gfx1250) — matrix operand order repaired and checked by reading, not compiled, not run on hardware
//
#include <hip/hip_runtime.h>
#include <stdint.h>
#include <stddef.h>

// ---------------------------------------------------------------------------
// 2-layer GRU (PyTorch gate order r,z,n), fused single-kernel recurrence.
// B=256, T=512, OBS=64, H=256.  Output: final hidden of layer 1, [B,H] f32.
//
// MI455X / gfx1250 (wave32, WMMA) strategy:
//  * Recurrence independent across batch -> 1 workgroup per 16-batch tile,
//    16 workgroups x 512 threads (16 waves). Whole T loop inside the kernel.
//  * Wave w owns H-columns [16w,16w+16) + its r/z/n gate rows. Gate math and
//    the h update happen in the WMMA f32 accumulator layout; h_prev stays in
//    VGPRs. Full h double-buffered in LDS (bf16) for next step's A-matrix and
//    for layer-1's input GEMM.
//  * Weights converted once fp32->bf16 into d_ws (~1.25MB, L2-resident).
//  * obs[:,t+1,:] prefetched into registers at top of step t; converted to
//    LDS after layer-1 WMMAs -> HBM latency hidden under matrix work.
//  * 2 barriers per step; t-loop unrolled x2 so buffer indices are constants.
// ---------------------------------------------------------------------------

typedef __bf16 bf16_t;
typedef __attribute__((ext_vector_type(16))) __bf16 v16bf;
typedef __attribute__((ext_vector_type(8)))  float  v8f;

#define GRU_B   256
#define GRU_T   512
#define GRU_OBS 64
#define GRU_H   256
#define TILE_B  16
#define NWAVE   16
#define NTHREADS (NWAVE * 32)
#define H_LD    264   // padded LDS row stride (elements), mult of 8, bank-skewed
#define OBS_LD  72

__device__ __forceinline__ bf16_t f2bf(float f) {
  union { float f; uint32_t u; } v; v.f = f;
  uint32_t r = v.u + 0x7fffu + ((v.u >> 16) & 1u);   // round-to-nearest-even
  union { uint16_t u; bf16_t b; } o; o.u = (uint16_t)(r >> 16);
  return o.b;
}

__device__ __forceinline__ float sigmoidf_(float x) {
  return 1.0f / (1.0f + __expf(-x));
}
__device__ __forceinline__ float tanhf_(float x) {
  return 2.0f / (1.0f + __expf(-2.0f * x)) - 1.0f;
}

// A-matrix fragment (16x32 bf16) from row-major LDS tile, origin (0, k0).
// lane<16 holds K {0..15}, lane>=16 holds K {16..31} (matches load_B).
__device__ __forceinline__ v16bf load_A(const bf16_t* S, int lds, int k0, int lane) {
  const int row = lane & 15;
  const int kb  = (lane >> 4) << 4;          // 0 or 16
  const bf16_t* p = S + row * lds + k0 + kb;
  union { v16bf v; uint4 q[2]; } u;
  u.q[0] = *(const uint4*)(p);
  u.q[1] = *(const uint4*)(p + 8);
  return u.v;
}

// B-matrix fragment (32x16 bf16). B column n == row (rowbase+n) of W (W is
// [3H,K] row-major; we multiply by W^T). lane<16: K=k0..k0+15, lane>=16:
// K=k0+16..k0+31, column = lane&15.  32 contiguous bytes per lane.
__device__ __forceinline__ v16bf load_B(const bf16_t* __restrict__ W, int ldw,
                                        int rowbase, int k0, int lane) {
  const int col = lane & 15;
  const int kb  = (lane >> 4) << 4;          // 0 or 16
  const bf16_t* p = W + (size_t)(rowbase + col) * ldw + k0 + kb;
  union { v16bf v; uint4 q[2]; } u;
  const uint4* q = (const uint4*)p;
  u.q[0] = q[0];
  u.q[1] = q[1];
  return u.v;
}

__device__ __forceinline__ v8f wmma_bf16(v16bf a, v16bf b, v8f c) {
  return __builtin_amdgcn_wmma_f32_16x16x32_bf16(false, a, false, b,
                                                 (short)0, c, false, false);
}

// Store a wave's 16x16 f32 tile (in C/D layout) into row-major bf16 LDS.
__device__ __forceinline__ void store_tile(bf16_t* S, int lds, int c16,
                                           int lane, v8f h) {
  const int col   = c16 + (lane & 15);
  const int mbase = (lane >> 4) << 3;        // rows 0..7 or 8..15
#pragma unroll
  for (int i = 0; i < 8; ++i) S[(mbase + i) * lds + col] = f2bf(h[i]);
}

__global__ void cvt_f32_bf16(const float* __restrict__ s, bf16_t* __restrict__ d, int n) {
  int i = blockIdx.x * blockDim.x + threadIdx.x;
  if (i < n) d[i] = f2bf(s[i]);
}

__global__ __launch_bounds__(NTHREADS)
void gru2_fused(const float* __restrict__ obs,
                const bf16_t* __restrict__ wih0, const bf16_t* __restrict__ whh0,
                const bf16_t* __restrict__ wih1, const bf16_t* __restrict__ whh1,
                const float* __restrict__ bih0, const float* __restrict__ bhh0,
                const float* __restrict__ bih1, const float* __restrict__ bhh1,
                float* __restrict__ out) {
  __shared__ bf16_t h0S[2][TILE_B][H_LD];    // layer-0 hidden, double buffered
  __shared__ bf16_t h1S[2][TILE_B][H_LD];    // layer-1 hidden, double buffered
  __shared__ bf16_t obsS[2][TILE_B][OBS_LD]; // obs[:,t,:] tile, double buffered

  const int tid   = threadIdx.x;
  const int lane  = tid & 31;
  const int w     = tid >> 5;                // wave id == column tile id
  const int c16   = w * 16;
  const int b0    = blockIdx.x * TILE_B;
  const int coln  = c16 + (lane & 15);       // this lane's H column (D layout)
  const int mbase = (lane >> 4) << 3;        // this lane's row base (D layout)

  // obs prefetch coordinates: thread covers (r0, k) and (r0+8, k)
  const int pr0 = tid >> 6;                  // 0..7
  const int pk  = tid & 63;

  // Per-lane gate biases (r,z merged x+h bias; n split: tanh(xn + r*hn)).
  const float br0  = bih0[coln]            + bhh0[coln];
  const float bz0  = bih0[GRU_H + coln]    + bhh0[GRU_H + coln];
  const float bxn0 = bih0[2 * GRU_H + coln];
  const float bhn0 = bhh0[2 * GRU_H + coln];
  const float br1  = bih1[coln]            + bhh1[coln];
  const float bz1  = bih1[GRU_H + coln]    + bhh1[GRU_H + coln];
  const float bxn1 = bih1[2 * GRU_H + coln];
  const float bhn1 = bhh1[2 * GRU_H + coln];

  // h(t=0) = 0; stage obs[:,0,:]
  for (int i = tid; i < TILE_B * H_LD; i += NTHREADS) {
    (&h0S[0][0][0])[i] = f2bf(0.0f);
    (&h1S[0][0][0])[i] = f2bf(0.0f);
  }
  obsS[0][pr0][pk]     = f2bf(obs[((size_t)(b0 + pr0)     * GRU_T + 0) * GRU_OBS + pk]);
  obsS[0][pr0 + 8][pk] = f2bf(obs[((size_t)(b0 + pr0 + 8) * GRU_T + 0) * GRU_OBS + pk]);
  v8f h0p = {}, h1p = {};
  __syncthreads();

#pragma unroll 2
  for (int t = 0; t < GRU_T; ++t) {
    const int cur = t & 1, nxt = cur ^ 1;

    // ---- issue obs[:,t+1,:] prefetch (waited on only after layer-1) ------
    float p0 = 0.0f, p1 = 0.0f;
    const bool have_next = (t + 1 < GRU_T);
    if (have_next) {
      p0 = obs[((size_t)(b0 + pr0)     * GRU_T + (t + 1)) * GRU_OBS + pk];
      p1 = obs[((size_t)(b0 + pr0 + 8) * GRU_T + (t + 1)) * GRU_OBS + pk];
    }

    // ---- layer 0: acc = x_t W_ih0^T + h0 W_hh0^T -------------------------
    v8f acr = {}, acz = {}, axn = {}, ahn = {};
#pragma unroll
    for (int k0 = 0; k0 < GRU_OBS; k0 += 32) {
      v16bf a = load_A(&obsS[cur][0][0], OBS_LD, k0, lane);
      acr = wmma_bf16(a, load_B(wih0, GRU_OBS, c16,             k0, lane), acr);
      acz = wmma_bf16(a, load_B(wih0, GRU_OBS, GRU_H + c16,     k0, lane), acz);
      axn = wmma_bf16(a, load_B(wih0, GRU_OBS, 2 * GRU_H + c16, k0, lane), axn);
    }
#pragma unroll 4
    for (int k0 = 0; k0 < GRU_H; k0 += 32) {
      v16bf a = load_A(&h0S[cur][0][0], H_LD, k0, lane);
      acr = wmma_bf16(a, load_B(whh0, GRU_H, c16,             k0, lane), acr);
      acz = wmma_bf16(a, load_B(whh0, GRU_H, GRU_H + c16,     k0, lane), acz);
      ahn = wmma_bf16(a, load_B(whh0, GRU_H, 2 * GRU_H + c16, k0, lane), ahn);
    }
    v8f hn0;
#pragma unroll
    for (int i = 0; i < 8; ++i) {
      float r = sigmoidf_(acr[i] + br0);
      float z = sigmoidf_(acz[i] + bz0);
      float n = tanhf_(axn[i] + bxn0 + r * (ahn[i] + bhn0));
      hn0[i] = n + z * (h0p[i] - n);         // (1-z)n + z h
    }
    h0p = hn0;
    store_tile(&h0S[nxt][0][0], H_LD, c16, lane, hn0);
    __syncthreads();                         // y0_t complete for all waves

    // ---- layer 1: acc = y0_t W_ih1^T + h1 W_hh1^T ------------------------
    v8f acr1 = {}, acz1 = {}, axn1 = {}, ahn1 = {};
#pragma unroll 4
    for (int k0 = 0; k0 < GRU_H; k0 += 32) {
      v16bf a = load_A(&h0S[nxt][0][0], H_LD, k0, lane);
      acr1 = wmma_bf16(a, load_B(wih1, GRU_H, c16,             k0, lane), acr1);
      acz1 = wmma_bf16(a, load_B(wih1, GRU_H, GRU_H + c16,     k0, lane), acz1);
      axn1 = wmma_bf16(a, load_B(wih1, GRU_H, 2 * GRU_H + c16, k0, lane), axn1);
    }
#pragma unroll 4
    for (int k0 = 0; k0 < GRU_H; k0 += 32) {
      v16bf a = load_A(&h1S[cur][0][0], H_LD, k0, lane);
      acr1 = wmma_bf16(a, load_B(whh1, GRU_H, c16,             k0, lane), acr1);
      acz1 = wmma_bf16(a, load_B(whh1, GRU_H, GRU_H + c16,     k0, lane), acz1);
      ahn1 = wmma_bf16(a, load_B(whh1, GRU_H, 2 * GRU_H + c16, k0, lane), ahn1);
    }
    v8f hn1;
#pragma unroll
    for (int i = 0; i < 8; ++i) {
      float r = sigmoidf_(acr1[i] + br1);
      float z = sigmoidf_(acz1[i] + bz1);
      float n = tanhf_(axn1[i] + bxn1 + r * (ahn1[i] + bhn1));
      hn1[i] = n + z * (h1p[i] - n);
    }
    h1p = hn1;
    store_tile(&h1S[nxt][0][0], H_LD, c16, lane, hn1);

    // ---- land the obs[:,t+1,:] prefetch into the other obs buffer --------
    if (have_next) {
      obsS[nxt][pr0][pk]     = f2bf(p0);
      obsS[nxt][pr0 + 8][pk] = f2bf(p1);
    }
    __syncthreads();
  }

  // ---- write final layer-1 hidden state: out[B,H] fp32 -------------------
#pragma unroll
  for (int i = 0; i < 8; ++i)
    out[(size_t)(b0 + mbase + i) * GRU_H + coln] = h1p[i];
}

extern "C" void kernel_launch(void* const* d_in, const int* in_sizes, int n_in,
                              void* d_out, int out_size, void* d_ws, size_t ws_size,
                              hipStream_t stream) {
  const float* obs   = (const float*)d_in[0];
  const float* wih0f = (const float*)d_in[1];
  const float* whh0f = (const float*)d_in[2];
  const float* bih0  = (const float*)d_in[3];
  const float* bhh0  = (const float*)d_in[4];
  const float* wih1f = (const float*)d_in[5];
  const float* whh1f = (const float*)d_in[6];
  const float* bih1  = (const float*)d_in[7];
  const float* bhh1  = (const float*)d_in[8];

  const int n_ih0 = 3 * GRU_H * GRU_OBS;   // 49152
  const int n_hh  = 3 * GRU_H * GRU_H;     // 196608

  bf16_t* wih0 = (bf16_t*)d_ws;
  bf16_t* whh0 = wih0 + n_ih0;
  bf16_t* wih1 = whh0 + n_hh;
  bf16_t* whh1 = wih1 + n_hh;

  cvt_f32_bf16<<<(n_ih0 + 255) / 256, 256, 0, stream>>>(wih0f, wih0, n_ih0);
  cvt_f32_bf16<<<(n_hh  + 255) / 256, 256, 0, stream>>>(whh0f, whh0, n_hh);
  cvt_f32_bf16<<<(n_hh  + 255) / 256, 256, 0, stream>>>(wih1f, wih1, n_hh);
  cvt_f32_bf16<<<(n_hh  + 255) / 256, 256, 0, stream>>>(whh1f, whh1, n_hh);

  gru2_fused<<<GRU_B / TILE_B, NTHREADS, 0, stream>>>(
      obs, wih0, whh0, wih1, whh1, bih0, bhh0, bih1, bhh1, (float*)d_out);
}
